// MHA_9955734192763
// MI455X (gfx1250) — compile-verified, matrix operand order checked by reading, not run on hardware
//
#include <hip/hip_runtime.h>
#include <hip/hip_bf16.h>
#include <cstdint>

// Problem constants (match reference)
#define Bsz 2
#define Tsz 2048
#define Csz 2048
#define Hn  16
#define HSd 128
#define ROTd 64
#define BT  (Bsz * Tsz)   // 4096

typedef __attribute__((ext_vector_type(16))) __bf16 v16bf;
typedef __attribute__((ext_vector_type(8)))  float  v8f;

union FragU { v16bf v; uint4 q[2]; };

static __device__ __forceinline__ unsigned short f2bf(float f) {
    union { float f; uint32_t u; } a; a.f = f;
    uint32_t r = a.u + 0x7FFFu + ((a.u >> 16) & 1u);
    return (unsigned short)(r >> 16);
}
static __device__ __forceinline__ float bf2f(unsigned short h) {
    union { uint32_t u; float f; } a; a.u = ((uint32_t)h) << 16;
    return a.f;
}

// Load a 16-bf16 WMMA operand fragment: lane&15 selects row (A) / col (B),
// lane>>4 selects the K sub-chunk per the CDNA5 16-bit operand layout.
static __device__ __forceinline__ v16bf frag_ld(const unsigned short* p) {
    FragU u;
    u.q[0] = *reinterpret_cast<const uint4*>(p);
    u.q[1] = *reinterpret_cast<const uint4*>(p + 16);
    return u.v;
}

// ---- CDNA5 async memory->LDS path (ASYNCcnt-tracked, no VGPR data) ------
// lds_addr: wave-relative LDS byte address (= low 32 bits of flat address,
// per ISA).  Each active lane moves 16 bytes.
static __device__ __forceinline__ void async_cp16(uint32_t lds_addr,
                                                  const unsigned short* g) {
    asm volatile("global_load_async_to_lds_b128 %0, %1, off"
                 :: "v"(lds_addr), "v"((unsigned long long)(uintptr_t)g)
                 : "memory");
}
static __device__ __forceinline__ void async_wait0() {
    asm volatile("s_wait_asynccnt 0x0" ::: "memory");
}

// ---------------------------------------------------------------- convert
__global__ void cvt_f32_bf16_k(const float* __restrict__ in,
                               unsigned short* __restrict__ out, int n) {
    for (int i = blockIdx.x * blockDim.x + threadIdx.x; i < n;
         i += gridDim.x * blockDim.x)
        out[i] = f2bf(in[i]);
}

// ---------------------------------------------------------------- GEMM
// Y[m,n] = sum_k A[m,k] * W[n,k] + bias[n]
// 128x128 block tile, 8 waves (2x4), BK=32, async double-buffered LDS.
// MODE 0: write bf16 into [B,H,T,HS] layout (QKV). MODE 1: write f32 [BT,C].
template <int MODE>
__global__ __launch_bounds__(256)
void gemm_wmma_k(const unsigned short* __restrict__ A,
                 const unsigned short* __restrict__ W,
                 const float* __restrict__ bias,
                 unsigned short* __restrict__ outB,
                 float* __restrict__ outF,
                 int M, int K, int N) {
    constexpr int BM = 128, BN = 128, BK = 32;
    __shared__ __align__(16) unsigned short sA[2][BM * BK];
    __shared__ __align__(16) unsigned short sB[2][BN * BK];

    const int tid  = threadIdx.x;
    const int lane = tid & 31, wave = tid >> 5;
    const int half = lane >> 4, l15 = lane & 15;
    const int wm = wave >> 2, wn = wave & 3;      // 2x4 wave grid
    const int bm = blockIdx.y * BM, bn = blockIdx.x * BN;

    const uint32_t aLds = (uint32_t)(uintptr_t)&sA[0][0];
    const uint32_t bLds = (uint32_t)(uintptr_t)&sB[0][0];

    v8f acc[4][2];
    for (int i = 0; i < 4; ++i)
        for (int n = 0; n < 2; ++n)
            for (int j = 0; j < 8; ++j) acc[i][n][j] = 0.f;

    // Issue one 128x32 bf16 tile (8 KB = 512 x 16B chunks, 2 per wave-lane set)
    auto stage = [&](const unsigned short* G, uint32_t base, int row0, int kb) {
        for (int j = 0; j < 2; ++j) {
            int id = (wave * 2 + j) * 32 + lane;          // 0..511
            int r = id >> 2, co = (id & 3) * 8;           // 4 chunks per row
            async_cp16(base + (uint32_t)(r * BK + co) * 2,
                       G + (size_t)(row0 + r) * K + kb + co);
        }
    };

    stage(A, aLds, bm, 0);
    stage(W, bLds, bn, 0);

    for (int kb = 0, it = 0; kb < K; kb += BK, ++it) {
        const int cur = it & 1;
        async_wait0();           // current tile's async copies done (this wave)
        __syncthreads();         // publish LDS across waves; prior reads done
        if (kb + BK < K) {       // overlap next tile's copy with compute
            stage(A, aLds + (1 - cur) * BM * BK * 2, bm, kb + BK);
            stage(W, bLds + (1 - cur) * BN * BK * 2, bn, kb + BK);
        }
        const unsigned short* cA = &sA[cur][0];
        const unsigned short* cB = &sB[cur][0];
        v16bf bf0 = frag_ld(&cB[(wn * 32 + l15) * BK + 8 * half]);
        v16bf bf1 = frag_ld(&cB[(wn * 32 + 16 + l15) * BK + 8 * half]);
        for (int mf = 0; mf < 4; ++mf) {
            v16bf af = frag_ld(&cA[(wm * 64 + mf * 16 + l15) * BK + 8 * half]);
            acc[mf][0] = __builtin_amdgcn_wmma_f32_16x16x32_bf16(
                false, af, false, bf0, (short)0, acc[mf][0], false, false);
            acc[mf][1] = __builtin_amdgcn_wmma_f32_16x16x32_bf16(
                false, af, false, bf1, (short)0, acc[mf][1], false, false);
        }
    }

    // epilogue: bias + layout transform
    for (int mf = 0; mf < 4; ++mf) {
        for (int nf = 0; nf < 2; ++nf) {
            for (int i = 0; i < 8; ++i) {
                int gm = bm + wm * 64 + mf * 16 + i + 8 * half;  // token row
                int gn = bn + wn * 32 + nf * 16 + l15;           // channel
                float v = acc[mf][nf][i] + bias[gn];
                if (MODE == 0) {
                    int b = gm >> 11, t = gm & (Tsz - 1);
                    int h = gn >> 7,  hs = gn & (HSd - 1);
                    outB[(((size_t)(b * Hn + h)) * Tsz + t) * HSd + hs] = f2bf(v);
                } else {
                    outF[(size_t)gm * N + gn] = v;
                }
            }
        }
    }
}

// ---------------------------------------------------------------- RoPE
// buf layout [B*H, T, HS]; rotate pairs (i, i+32) of first 64 dims.
__global__ void rope_k(unsigned short* __restrict__ buf) {
    int tid = blockIdx.x * blockDim.x + threadIdx.x;
    if (tid >= Bsz * Hn * Tsz * (ROTd / 2)) return;
    int i  = tid & 31;
    int t  = (tid >> 5) & (Tsz - 1);
    int bh = tid >> 16;
    size_t base = ((size_t)bh * Tsz + t) * HSd;
    // inv_freq = 10000^(-2i/64) = exp(-i * ln(10000)/32)
    float inv = __expf(-(float)i * 0.2878231366f);
    float ang = (float)t * inv;
    float s, c;
    __sincosf(ang, &s, &c);
    float a = bf2f(buf[base + i]);
    float b = bf2f(buf[base + i + 32]);
    buf[base + i]      = f2bf(a * c - b * s);
    buf[base + i + 32] = f2bf(b * c + a * s);
}

// ---------------------------------------------------------------- attention
// One block = (b,h, 64 q-rows). 4 waves, each owns 16 q-rows × full HS=128.
// Q staged in LDS (async) to keep the wave under 256 VGPRs.
__global__ __launch_bounds__(128)
void attn_wmma_k(const unsigned short* __restrict__ Q,
                 const unsigned short* __restrict__ Kt,
                 const unsigned short* __restrict__ V,
                 unsigned short* __restrict__ Oatt) {
    __shared__ __align__(16) unsigned short sQ[64 * HSd];    // [q][d]
    __shared__ __align__(16) unsigned short sK[64 * HSd];    // [k][d]
    __shared__ __align__(16) unsigned short sV[HSd * 64];    // [d][k] transposed
    __shared__ __align__(16) unsigned short sP[4][16 * 64];  // per-wave P tile

    const int tid  = threadIdx.x;
    const int lane = tid & 31, wave = tid >> 5;
    const int half = lane >> 4, l15 = lane & 15;
    const int q0 = blockIdx.x * 64;
    const int bh = blockIdx.y;
    const size_t hbase = (size_t)bh * Tsz * HSd;
    const uint32_t qLds = (uint32_t)(uintptr_t)&sQ[0];
    const uint32_t kLds = (uint32_t)(uintptr_t)&sK[0];

    // Stage the 64x128 Q tile once via async copy (1024 x 16B chunks)
    for (int j = 0; j < 8; ++j) {
        int id = (wave * 8 + j) * 32 + lane;
        int r = id >> 4, d0 = (id & 15) * 8;
        async_cp16(qLds + (uint32_t)(r * HSd + d0) * 2,
                   Q + hbase + (size_t)(q0 + r) * HSd + d0);
    }

    v8f o[8];
    for (int d = 0; d < 8; ++d)
        for (int i = 0; i < 8; ++i) o[d][i] = 0.f;
    float mi[8], li[8];
    for (int i = 0; i < 8; ++i) { mi[i] = -3.0e38f; li[i] = 0.f; }

    const float scale = 0.08838834764831845f;  // 1/sqrt(128)
    const int rowg0 = q0 + wave * 16 + 8 * half;

    for (int kt = 0; kt <= (int)blockIdx.x; ++kt) {
        const int k0 = kt * 64;
        // K tile via async memory->LDS (64x128 bf16 = 1024 x 16B chunks)
        for (int j = 0; j < 8; ++j) {
            int id = (wave * 8 + j) * 32 + lane;          // 0..1023
            int r = id >> 4, d0 = (id & 15) * 8;          // 16 chunks per row
            async_cp16(kLds + (uint32_t)(r * HSd + d0) * 2,
                       Kt + hbase + (size_t)(k0 + r) * HSd + d0);
        }
        // V tile staged transposed (manual: needs d<->k transpose)
        #pragma unroll 2
        for (int c = tid; c < 1024; c += 128) {
            int r = c >> 4, d0 = (c & 15) * 8;
            union { uint4 q; unsigned short h[8]; } uu;
            uu.q = *reinterpret_cast<const uint4*>(&V[hbase + (size_t)(k0 + r) * HSd + d0]);
            for (int j = 0; j < 8; ++j) sV[(d0 + j) * 64 + r] = uu.h[j];
        }
        async_wait0();
        __syncthreads();

        // S = Q K^T  (16 x 64): ks outer so the Q fragment is transient
        v8f s4[4];
        for (int nf = 0; nf < 4; ++nf)
            for (int j = 0; j < 8; ++j) s4[nf][j] = 0.f;
        for (int ks = 0; ks < 4; ++ks) {
            v16bf qf = frag_ld(&sQ[(wave * 16 + l15) * HSd + ks * 32 + 8 * half]);
            for (int nf = 0; nf < 4; ++nf) {
                v16bf bfr = frag_ld(&sK[(nf * 16 + l15) * HSd + ks * 32 + 8 * half]);
                s4[nf] = __builtin_amdgcn_wmma_f32_16x16x32_bf16(
                    false, qf, false, bfr, (short)0, s4[nf], false, false);
            }
        }

        // online softmax (rows live across 16-lane halves per C/D layout)
        for (int i = 0; i < 8; ++i) {
            float rm = -3.0e38f;
            for (int nf = 0; nf < 4; ++nf) {
                int colg = k0 + nf * 16 + l15;
                float sv = s4[nf][i] * scale;
                if (colg > rowg0 + i) sv = -3.0e38f;  // causal mask
                s4[nf][i] = sv;
                rm = fmaxf(rm, sv);
            }
            for (int off = 8; off >= 1; off >>= 1)
                rm = fmaxf(rm, __shfl_xor(rm, off, 32));
            float mn = fmaxf(mi[i], rm);
            float corr = __expf(mi[i] - mn);
            mi[i] = mn;
            float rs = 0.f;
            for (int nf = 0; nf < 4; ++nf) {
                float e = __expf(s4[nf][i] - mn);
                s4[nf][i] = e;
                rs += e;
            }
            for (int off = 8; off >= 1; off >>= 1)
                rs += __shfl_xor(rs, off, 32);
            li[i] = li[i] * corr + rs;
            for (int d = 0; d < 8; ++d) o[d][i] *= corr;
            int prow = i + 8 * half;
            for (int nf = 0; nf < 4; ++nf)
                sP[wave][prow * 64 + nf * 16 + l15] = f2bf(s4[nf][i]);
        }
        __syncthreads();

        // O += P @ V   (P: 16x64 -> 2 K-steps; V^T staged: contiguous B frags)
        for (int ks = 0; ks < 2; ++ks) {
            v16bf pa = frag_ld(&sP[wave][l15 * 64 + ks * 32 + 8 * half]);
            for (int d = 0; d < 8; ++d) {
                v16bf bv = frag_ld(&sV[(d * 16 + l15) * 64 + ks * 32 + 8 * half]);
                o[d] = __builtin_amdgcn_wmma_f32_16x16x32_bf16(
                    false, pa, false, bv, (short)0, o[d], false, false);
            }
        }
        __syncthreads();
    }

    // normalize + store to [BT, C] bf16 for the output projection
    const int b = bh >> 4, h = bh & 15;
    for (int d = 0; d < 8; ++d) {
        for (int i = 0; i < 8; ++i) {
            int t = q0 + wave * 16 + i + 8 * half;
            int col = h * HSd + d * 16 + l15;
            float val = o[d][i] / li[i];
            Oatt[((size_t)(b * Tsz + t)) * Csz + col] = f2bf(val);
        }
    }
}

// ---------------------------------------------------------------- launch
extern "C" void kernel_launch(void* const* d_in, const int* in_sizes, int n_in,
                              void* d_out, int out_size, void* d_ws, size_t ws_size,
                              hipStream_t stream) {
    const float* x  = (const float*)d_in[0];
    const float* Wq = (const float*)d_in[1];
    const float* bq = (const float*)d_in[2];
    const float* Wk = (const float*)d_in[3];
    const float* bk = (const float*)d_in[4];
    const float* Wv = (const float*)d_in[5];
    const float* bv = (const float*)d_in[6];
    const float* Wo = (const float*)d_in[7];
    const float* bo = (const float*)d_in[8];
    float* out = (float*)d_out;

    const size_t NX = (size_t)BT * Csz;      // 8.39M elems
    const size_t NW = (size_t)Csz * Csz;     // 4.19M elems
    unsigned short* p   = (unsigned short*)d_ws;
    unsigned short* xb  = p; p += NX;
    unsigned short* wqb = p; p += NW;
    unsigned short* wkb = p; p += NW;
    unsigned short* wvb = p; p += NW;
    unsigned short* wob = p; p += NW;
    unsigned short* Qb  = p; p += NX;        // [B,H,T,HS]
    unsigned short* Kb  = p; p += NX;
    unsigned short* Vb  = p; p += NX;
    unsigned short* Ab  = p; p += NX;        // attn result [BT,C]
    (void)ws_size; (void)n_in; (void)in_sizes; (void)out_size;

    // 1) fp32 -> bf16
    cvt_f32_bf16_k<<<4096, 256, 0, stream>>>(x,  xb,  (int)NX);
    cvt_f32_bf16_k<<<4096, 256, 0, stream>>>(Wq, wqb, (int)NW);
    cvt_f32_bf16_k<<<4096, 256, 0, stream>>>(Wk, wkb, (int)NW);
    cvt_f32_bf16_k<<<4096, 256, 0, stream>>>(Wv, wvb, (int)NW);
    cvt_f32_bf16_k<<<4096, 256, 0, stream>>>(Wo, wob, (int)NW);

    // 2) QKV projections (WMMA, async double-buffered)
    dim3 gg(Csz / 128, BT / 128);
    gemm_wmma_k<0><<<gg, 256, 0, stream>>>(xb, wqb, bq, Qb, nullptr, BT, Csz, Csz);
    gemm_wmma_k<0><<<gg, 256, 0, stream>>>(xb, wkb, bk, Kb, nullptr, BT, Csz, Csz);
    gemm_wmma_k<0><<<gg, 256, 0, stream>>>(xb, wvb, bv, Vb, nullptr, BT, Csz, Csz);

    // 3) RoPE on Q, K
    int nrope = Bsz * Hn * Tsz * (ROTd / 2);
    rope_k<<<(nrope + 255) / 256, 256, 0, stream>>>(Qb);
    rope_k<<<(nrope + 255) / 256, 256, 0, stream>>>(Kb);

    // 4) causal flash attention (WMMA + async Q/K staging)
    dim3 ga(Tsz / 64, Bsz * Hn);
    attn_wmma_k<<<ga, 128, 0, stream>>>(Qb, Kb, Vb, Ab);

    // 5) output projection, fp32 result
    gemm_wmma_k<1><<<gg, 256, 0, stream>>>(Ab, wob, bo, nullptr, out, BT, Csz, Csz);
}
